// Conv2d_28458453303904
// MI455X (gfx1250) — matrix operand order repaired and checked by reading, not compiled, not run on hardware
//
#include <hip/hip_runtime.h>

// ---- types ----
typedef _Float16 v4h  __attribute__((ext_vector_type(4)));
typedef _Float16 v8h  __attribute__((ext_vector_type(8)));
typedef _Float16 v16h __attribute__((ext_vector_type(16)));
typedef float    v8f  __attribute__((ext_vector_type(8)));

typedef unsigned int uint32x4 __attribute__((ext_vector_type(4)));
typedef int          int32x8  __attribute__((ext_vector_type(8)));
typedef int          int32x4  __attribute__((ext_vector_type(4)));

#define CIN   128
#define COUT  256
#define HW    256
#define HWP   258              // padded spatial extent (1-px zero border)
#define KTAPS 9
#define KDIM  (KTAPS * CIN)    // 1152

// ---------------------------------------------------------------------------
// Prep kernel 1: x (CHW fp32) -> zero-padded HWC f16: xp[258][258][128].
// Channel-fastest output => coalesced writes; padding stored as real zeros so
// the TDM tile loads need no predication.
// ---------------------------------------------------------------------------
__global__ __launch_bounds__(256) void pad_x_kernel(const float* __restrict__ x,
                                                    _Float16* __restrict__ xp) {
  size_t i = (size_t)blockIdx.x * 256 + threadIdx.x;   // < 258*258*128
  int c    = (int)(i & (CIN - 1));
  int rest = (int)(i >> 7);
  int xx   = rest % HWP;
  int yy   = rest / HWP;
  _Float16 v = (_Float16)0.f;
  if (yy >= 1 && yy <= HW && xx >= 1 && xx <= HW)
    v = (_Float16)x[(size_t)c * (HW * HW) + (size_t)(yy - 1) * HW + (xx - 1)];
  xp[i] = v;
}

// ---------------------------------------------------------------------------
// Prep kernel 2: weight (cout, cin, 3, 3) fp32 -> wt[cout][tap][cin] f16
// (tap-major K ordering: k = t*128 + c)
// ---------------------------------------------------------------------------
__global__ __launch_bounds__(256) void pack_w_kernel(const float* __restrict__ w,
                                                     _Float16* __restrict__ wt) {
  int idx = blockIdx.x * 256 + threadIdx.x;   // < COUT*KDIM = 294912
  int m   = idx / KDIM;
  int rem = idx - m * KDIM;
  int t   = rem >> 7;
  int c   = rem & 127;
  wt[idx] = (_Float16)w[(m * CIN + c) * KTAPS + t];
}

// ---------------------------------------------------------------------------
// Main kernel: implicit-GEMM conv, TDM-staged double-buffered LDS slabs,
// v_wmma_f32_16x16x32_f16 compute.
//
// Block: 256 threads (8 waves). Block tile: 128 cout x 32 pixels (one row y).
// Loop: 4 channel-chunks (TDM slab each) x 9 taps (shifted LDS windows).
// ---------------------------------------------------------------------------
__global__ __launch_bounds__(256) void conv_wmma_kernel(
    const _Float16* __restrict__ xp,   // [258][258][128] f16 (padded HWC)
    const _Float16* __restrict__ wt,   // [COUT][9][128] f16
    const float*    __restrict__ bias, // [COUT]
    float*          __restrict__ out)  // [COUT][256][256]
{
  // Double-buffered input slab: [buf][row 0..2][col 0..33][ch 0..31]
  // 32 contiguous bytes per (row,col,ch-group) => two ds_load_b128 per B-frag.
  __shared__ __align__(16) _Float16 S[2][3][34][32];

  const int tid  = threadIdx.x;
  const int wave = tid >> 5;
  const int lane = tid & 31;
  const int lh   = lane & 15;      // lane % 16 -> N column / M row within tile
  const int hi   = lane >> 4;      // 0: lanes 0-15, 1: lanes 16-31

  const int y  = blockIdx.x >> 3;               // output row
  const int x0 = (blockIdx.x & 7) * 32;         // first output col of tile
  const int m0 = blockIdx.y * 128 + wave * 16;  // first cout row for this wave

  // TDM issue: 3-D tile [ch=32][col=34][row=3] from padded HWC tensor to LDS.
  // D# per CDNA5 ISA ch.8: count=1, type=2(image), data_size=1(2B),
  // dim0 stride(col step)=128, dim1 stride(row step)=258*128=33024.
  auto tdm_issue = [&](int buf, int c0) {
    unsigned lds = (unsigned)(unsigned long long)(void*)&S[buf][0][0][0];
    unsigned long long ga = (unsigned long long)(const void*)xp +
        2ull * ((((unsigned long long)y) * HWP + (unsigned)x0) * CIN + (unsigned)c0);
    uint32x4 g0 = { 1u,                                   // count=1 (valid D#)
                    lds,                                  // lds_addr
                    (unsigned)ga,                         // global_addr[31:0]
                    (unsigned)(ga >> 32) | (2u << 30) };  // ga[56:32] | type=2
    int32x8 g1 = { (int)(1u << 16),          // data_size=1 (2 bytes)
                   (int)(128u << 16),        // tensor_dim0 = 128 (channels)
                   (int)(258u << 16),        // tensor_dim1 = 258 (cols)
                   (int)(32u << 16),         // tile_dim0 = 32 (channels)
                   (int)(34u | (3u << 16)),  // tile_dim1 = 34, tile_dim2 = 3
                   128,                      // tensor_dim0_stride = 128
                   (int)(33024u << 16),      // tensor_dim1_stride = 33024 (lo16)
                   0 };                      // tensor_dim1_stride hi = 0
    int32x4 g2 = { HWP, 1, 0, 0 };           // tensor_dim2 = 258; dim3 unused
    int32x4 g3 = { 0, 0, 0, 0 };
    int32x8 g4 = { 0, 0, 0, 0, 0, 0, 0, 0 }; // extra group (clang-23 6-arg form)
    __builtin_amdgcn_tensor_load_to_lds(g0, g1, g2, g3, g4, 0);
  };

  v8f acc0 = {0.f, 0.f, 0.f, 0.f, 0.f, 0.f, 0.f, 0.f};
  v8f acc1 = {0.f, 0.f, 0.f, 0.f, 0.f, 0.f, 0.f, 0.f};

  // A-fragment base (16-bit A 16x32 layout): lanes 0-15: M=lane, K={0..7,16..23};
  // lanes 16-31: M=lane-16, K={8..15,24..31}.
  const _Float16* arow = wt + (size_t)(m0 + lh) * KDIM + hi * 8;

  if (wave == 0) tdm_issue(0, 0);   // prologue: slab for chunk 0

#pragma unroll
  for (int cc = 0; cc < 4; ++cc) {
    const int c0 = cc << 5;
    if (wave == 0) {
      if (cc < 3) {
        tdm_issue((cc + 1) & 1, (cc + 1) << 5);  // prefetch next slab
        __builtin_amdgcn_s_wait_tensorcnt(1);    // wait for slab(cc) only
      } else {
        __builtin_amdgcn_s_wait_tensorcnt(0);    // last chunk: drain
      }
    }
    __syncthreads();   // slab(cc) visible to all waves

    const _Float16* slab = &S[cc & 1][0][0][0];

#pragma unroll
    for (int t = 0; t < KTAPS; ++t) {
      const int dy = t / 3;   // 0..2 (padded row offset)
      const int dx = t % 3;   // 0..2 (padded col offset)

      // ---- A fragment: two 16B global loads ----
      const int k0 = t * CIN + c0;
      const v8h alo = *(const v8h*)(arow + k0);
      const v8h ahi = *(const v8h*)(arow + k0 + 16);
      const v16h a = __builtin_shufflevector(alo, ahi,
          0, 1, 2, 3, 4, 5, 6, 7, 8, 9, 10, 11, 12, 13, 14, 15);

      // ---- B fragments: shifted windows of the slab, two ds_load_b128 ----
      const _Float16* bp0 = slab + ((dy * 34) + (lh + dx)) * 32 + hi * 8;
      const _Float16* bp1 = bp0 + 16 * 32;   // columns n = 16 + lh
      const v8h b0lo = *(const v8h*)(bp0);
      const v8h b0hi = *(const v8h*)(bp0 + 16);
      const v8h b1lo = *(const v8h*)(bp1);
      const v8h b1hi = *(const v8h*)(bp1 + 16);
      const v16h b0 = __builtin_shufflevector(b0lo, b0hi,
          0, 1, 2, 3, 4, 5, 6, 7, 8, 9, 10, 11, 12, 13, 14, 15);
      const v16h b1 = __builtin_shufflevector(b1lo, b1hi,
          0, 1, 2, 3, 4, 5, 6, 7, 8, 9, 10, 11, 12, 13, 14, 15);

      acc0 = __builtin_amdgcn_wmma_f32_16x16x32_f16(
          false, a, false, b0, (short)0, acc0, false, false);
      acc1 = __builtin_amdgcn_wmma_f32_16x16x32_f16(
          false, a, false, b1, (short)0, acc1, false, false);
    }
    __syncthreads();   // slab(cc) consumed; buffer free for chunk cc+2
  }

  // ---- epilogue: bias add + store (C/D layout: VGPR r -> M = r + 8*hi, N = lh) ----
#pragma unroll
  for (int r = 0; r < 8; ++r) {
    const int m = m0 + r + hi * 8;
    const float bv = bias[m];
    float* orow = out + ((size_t)m * HW + y) * HW + x0;
    orow[lh]      = acc0[r] + bv;
    orow[16 + lh] = acc1[r] + bv;
  }
}

// ---------------------------------------------------------------------------
// Entry point
// ---------------------------------------------------------------------------
extern "C" void kernel_launch(void* const* d_in, const int* in_sizes, int n_in,
                              void* d_out, int out_size, void* d_ws, size_t ws_size,
                              hipStream_t stream) {
  (void)in_sizes; (void)n_in; (void)out_size; (void)ws_size;

  const float* x = (const float*)d_in[0];   // (128, 256, 256)
  const float* w = (const float*)d_in[1];   // (256, 128, 3, 3)
  const float* b = (const float*)d_in[2];   // (256, 1, 1)
  float* out = (float*)d_out;               // (256, 256, 256)

  char* ws = (char*)d_ws;
  _Float16* xp = (_Float16*)ws;                                   // 258*258*128 f16
  _Float16* wt = (_Float16*)(ws + (size_t)HWP * HWP * CIN * sizeof(_Float16));

  // Prep (same stream => ordered before main kernel)
  pad_x_kernel<<<(HWP * HWP * CIN) / 256, 256, 0, stream>>>(x, xp);
  pack_w_kernel<<<(COUT * KDIM) / 256, 256, 0, stream>>>(w, wt);

  // Main conv: grid.x = rows * (W/32) = 256*8, grid.y = COUT/128 = 2
  dim3 grid(HW * (HW / 32), COUT / 128);
  conv_wmma_kernel<<<grid, 256, 0, stream>>>(xp, wt, b, out);
}
